// Decoder_20074677141556
// MI455X (gfx1250) — matrix operand order repaired and checked by reading, not compiled, not run on hardware
//
#include <hip/hip_runtime.h>

typedef unsigned short u16;
typedef __attribute__((ext_vector_type(16))) __bf16 v16bf;
typedef __attribute__((ext_vector_type(8)))  float  v8f;
typedef __attribute__((ext_vector_type(8)))  unsigned short u16x8;

static constexpr int kE  = 512;
static constexpr int kH  = 512;
static constexpr int kB  = 64;
static constexpr int kT  = 32;
static constexpr int kV  = 32000;
static constexpr int kG  = 2048;   // 4*H
static constexpr int kBT = 2048;   // B*T

union AFrag { v16bf bf; u16x8 h[2]; };

__device__ __forceinline__ u16 f2bf(float f) {
  unsigned int u = __float_as_uint(f);
  u += 0x7FFFu + ((u >> 16) & 1u);   // round-to-nearest-even
  return (u16)(u >> 16);
}

// A-matrix 16x32 bf16 fragment, src row-major [M, ld] (K contiguous).
// ISA layout: lanes 0-15 hold K=0..7 (v0-3) and 16..23 (v4-7); lanes 16-31 hold K=8..15, 24..31.
__device__ __forceinline__ v16bf load_a16x32(const u16* A, int ld, int mbase, int kbase, int lane) {
  const int row = mbase + (lane & 15);
  const int off = (lane >> 4) * 8;
  const u16* p = A + row * ld + kbase + off;
  AFrag f;
  f.h[0] = *(const u16x8*)(p);
  f.h[1] = *(const u16x8*)(p + 16);
  return f.bf;
}

// B-matrix 32x16 bf16 fragment; source stored row-major as [N, ld] (i.e. B = srcT, K contiguous).
// ISA layout: lanes 0-15 hold K=0..7 and 16..23, lanes 16-31 hold K=8..15 and 24..31, column N = lane&15.
__device__ __forceinline__ v16bf load_b32x16(const u16* Bm, int ld, int nbase, int kbase, int lane) {
  const int row = nbase + (lane & 15);
  const int off = (lane >> 4) * 8;
  const u16* p = Bm + row * ld + kbase + off;
  AFrag f;
  f.h[0] = *(const u16x8*)(p);
  f.h[1] = *(const u16x8*)(p + 16);
  return f.bf;
}

__device__ __forceinline__ v8f wmma_bf16(v16bf a, v16bf b, v8f c) {
  return __builtin_amdgcn_wmma_f32_16x16x32_bf16(false, a, false, b, (short)0, c, false, false);
}

__device__ __forceinline__ float sigmoidf(float x) { return 1.0f / (1.0f + __expf(-x)); }

// ---------------- f32 -> bf16 weight conversion ----------------
__global__ void k_convert(const float* __restrict__ src, u16* __restrict__ dst, int n) {
  int i = blockIdx.x * blockDim.x + threadIdx.x;
  if (i < n) dst[i] = f2bf(src[i]);
}

// ---------------- build LSTM input x0 = [features ; emb[sentence[:, :-1]]] as bf16 [BT, E] ----
__global__ void k_build_x(const int* __restrict__ sentence, const float* __restrict__ features,
                          const float* __restrict__ emb, u16* __restrict__ x0) {
  const int bt = blockIdx.x;
  const int b = bt >> 5, t = bt & 31;
  const float* src = (t == 0) ? (features + b * kE)
                              : (emb + (size_t)sentence[b * kT + (t - 1)] * kE);
  for (int e = threadIdx.x; e < kE; e += blockDim.x)
    x0[bt * kE + e] = f2bf(src[e]);
}

// ---------------- xp[t][g][b] = bias_ih[g] + bias_hh[g] + (x @ W_ihT)  -----------------------
// M = g (4H), N = bt, K = E.  block=128 (4 waves), grid=(BT/16, (4H/16)/4)
__global__ void k_xp_gemm(const u16* __restrict__ x, const u16* __restrict__ W,
                          const float* __restrict__ b_ih, const float* __restrict__ b_hh,
                          float* __restrict__ xp) {
  const int lane = threadIdx.x & 31;
  const int wave = threadIdx.x >> 5;
  const int mbase = (blockIdx.y * 4 + wave) * 16;
  const int nbase = blockIdx.x * 16;
  const int sel = lane >> 4;

  v8f c;
#pragma unroll
  for (int r = 0; r < 8; ++r) {
    int g = mbase + r + sel * 8;
    c[r] = b_ih[g] + b_hh[g];
  }
#pragma unroll
  for (int kb = 0; kb < kE; kb += 32) {
    v16bf a = load_a16x32(W, kE, mbase, kb, lane);
    v16bf bb = load_b32x16(x, kE, nbase, kb, lane);
    c = wmma_bf16(a, bb, c);
  }
  const int bt = nbase + (lane & 15);
  const int b = bt >> 5, t = bt & 31;
#pragma unroll
  for (int r = 0; r < 8; ++r) {
    int g = mbase + r + sel * 8;
    xp[(t * kG + g) * kB + b] = c[r];
  }
}

// ---------------- pregates[g][b] = xp[t][g][b] + h_cur @ W_hhT --------------------------------
// M = g (4H, 128 tiles), N = b (64, 4 tiles), K = H.  block=128, grid=(4, 32)
__global__ void k_step_gemm(const float* __restrict__ xp, const u16* __restrict__ Whh,
                            const u16* __restrict__ h_cur, float* __restrict__ pre, int t) {
  const int lane = threadIdx.x & 31;
  const int wave = threadIdx.x >> 5;
  const int mbase = (blockIdx.y * 4 + wave) * 16;
  const int nbase = blockIdx.x * 16;
  const int sel = lane >> 4;
  const int ln = lane & 15;

  v8f c;
#pragma unroll
  for (int r = 0; r < 8; ++r) {
    int g = mbase + r + sel * 8;
    c[r] = xp[(t * kG + g) * kB + nbase + ln];
  }
  if (t > 0) {  // h == 0 at t==0 -> skip recurrent term
#pragma unroll
    for (int kb = 0; kb < kH; kb += 32) {
      v16bf a = load_a16x32(Whh, kH, mbase, kb, lane);
      v16bf bb = load_b32x16(h_cur, kH, nbase, kb, lane);
      c = wmma_bf16(a, bb, c);
    }
  }
#pragma unroll
  for (int r = 0; r < 8; ++r) {
    int g = mbase + r + sel * 8;
    pre[g * kB + nbase + ln] = c[r];
  }
}

// ---------------- LSTM cell pointwise update --------------------------------------------------
// 32768 threads: b = i&63 (contiguous pregate/c reads), j = i>>6.
__global__ void k_cell(const float* __restrict__ pre, float* __restrict__ c_state,
                       u16* __restrict__ h_cur, u16* __restrict__ h_seq, int t) {
  const int i = blockIdx.x * blockDim.x + threadIdx.x;
  const int b = i & (kB - 1);
  const int j = i >> 6;
  const float gi = sigmoidf(pre[(j          ) * kB + b]);
  const float gf = sigmoidf(pre[(j +     kH ) * kB + b]);
  const float gg = tanhf   (pre[(j + 2 * kH ) * kB + b]);
  const float go = sigmoidf(pre[(j + 3 * kH ) * kB + b]);
  const float cp = (t == 0) ? 0.0f : c_state[j * kB + b];
  const float cn = gf * cp + gi * gg;
  const float hn = go * tanhf(cn);
  c_state[j * kB + b] = cn;
  const u16 hb = f2bf(hn);
  h_cur[b * kH + j] = hb;                    // [B, H] for next-step GEMM B operand
  h_seq[(b * kT + t) * kH + j] = hb;         // [BT, H] sequence output
}

// ---------------- FC: out[b, v, t] = fc_W @ h1T + fc_b ---------------------------------------
// M = v (2000 tiles), N = bt, K = H.  Each wave: 2 M-tiles x 4 N-tiles
// (A fragments reused 4x, B fragments reused 2x -> 1.5 b128 loads per WMMA).
// block=256 (8 waves), grid=(BT/64 = 32, 2000/(8*2) = 125)
__global__ void k_fc(const u16* __restrict__ fcW, const u16* __restrict__ h_all,
                     const float* __restrict__ fc_b, float* __restrict__ out) {
  const int lane = threadIdx.x & 31;
  const int wave = threadIdx.x >> 5;
  const int mbase = (blockIdx.y * 8 + wave) * 32;   // 2 M-tiles: mbase, mbase+16
  const int ngrp = blockIdx.x * 4;
  const int sel = lane >> 4;
  const int ln = lane & 15;

  v8f acc[2][4] = {};
#pragma unroll
  for (int kb = 0; kb < kH; kb += 32) {
    v16bf a0 = load_a16x32(fcW, kH, mbase,      kb, lane);
    v16bf a1 = load_a16x32(fcW, kH, mbase + 16, kb, lane);
#pragma unroll
    for (int q = 0; q < 4; ++q) {
      v16bf bq = load_b32x16(h_all, kH, (ngrp + q) * 16, kb, lane);
      acc[0][q] = wmma_bf16(a0, bq, acc[0][q]);
      acc[1][q] = wmma_bf16(a1, bq, acc[1][q]);
    }
  }
#pragma unroll
  for (int mi = 0; mi < 2; ++mi) {
#pragma unroll
    for (int q = 0; q < 4; ++q) {
      const int bt = (ngrp + q) * 16 + ln;
      const int b = bt >> 5, t = bt & 31;
#pragma unroll
      for (int r = 0; r < 8; ++r) {
        const int v = mbase + mi * 16 + r + sel * 8;
        out[((size_t)b * kV + v) * kT + t] = acc[mi][q][r] + fc_b[v];
      }
    }
  }
}

extern "C" void kernel_launch(void* const* d_in, const int* in_sizes, int n_in,
                              void* d_out, int out_size, void* d_ws, size_t ws_size,
                              hipStream_t stream) {
  (void)in_sizes; (void)n_in; (void)out_size; (void)ws_size;
  const int*   sentence = (const int*)  d_in[0];
  const float* features = (const float*)d_in[1];
  /* lengths d_in[2] unused */
  const float* emb      = (const float*)d_in[3];
  const float* W_ih0    = (const float*)d_in[4];
  const float* W_hh0    = (const float*)d_in[5];
  const float* b_ih0    = (const float*)d_in[6];
  const float* b_hh0    = (const float*)d_in[7];
  const float* W_ih1    = (const float*)d_in[8];
  const float* W_hh1    = (const float*)d_in[9];
  const float* b_ih1    = (const float*)d_in[10];
  const float* b_hh1    = (const float*)d_in[11];
  const float* fc_W     = (const float*)d_in[12];
  const float* fc_b     = (const float*)d_in[13];
  float* out = (float*)d_out;

  uintptr_t p = (uintptr_t)d_ws;
  auto take = [&](size_t bytes) -> void* {
    uintptr_t r = (p + 255) & ~(uintptr_t)255;
    p = r + bytes;
    return (void*)r;
  };
  u16*   wih0 = (u16*)  take((size_t)kG * kE * 2);
  u16*   whh0 = (u16*)  take((size_t)kG * kH * 2);
  u16*   wih1 = (u16*)  take((size_t)kG * kH * 2);
  u16*   whh1 = (u16*)  take((size_t)kG * kH * 2);
  u16*   fcw  = (u16*)  take((size_t)kV * kH * 2);
  u16*   x0   = (u16*)  take((size_t)kBT * kE * 2);
  u16*   x1   = (u16*)  take((size_t)kBT * kH * 2);
  u16*   h1a  = (u16*)  take((size_t)kBT * kH * 2);
  float* xp   = (float*)take((size_t)kT * kG * kB * 4);
  float* pre  = (float*)take((size_t)kG * kB * 4);
  float* cs   = (float*)take((size_t)kH * kB * 4);
  u16*   hc   = (u16*)  take((size_t)kB * kH * 2);

  auto cvt = [&](const float* s, u16* d, int n) {
    k_convert<<<(n + 255) / 256, 256, 0, stream>>>(s, d, n);
  };
  cvt(W_ih0, wih0, kG * kE);
  cvt(W_hh0, whh0, kG * kH);
  cvt(W_ih1, wih1, kG * kH);
  cvt(W_hh1, whh1, kG * kH);
  cvt(fc_W,  fcw,  kV * kH);

  k_build_x<<<kBT, 256, 0, stream>>>(sentence, features, emb, x0);

  // Layer 0
  k_xp_gemm<<<dim3(kBT / 16, (kG / 16) / 4), 128, 0, stream>>>(x0, wih0, b_ih0, b_hh0, xp);
  for (int t = 0; t < kT; ++t) {
    k_step_gemm<<<dim3(kB / 16, (kG / 16) / 4), 128, 0, stream>>>(xp, whh0, hc, pre, t);
    k_cell<<<(kB * kH) / 256, 256, 0, stream>>>(pre, cs, hc, x1, t);
  }

  // Layer 1
  k_xp_gemm<<<dim3(kBT / 16, (kG / 16) / 4), 128, 0, stream>>>(x1, wih1, b_ih1, b_hh1, xp);
  for (int t = 0; t < kT; ++t) {
    k_step_gemm<<<dim3(kB / 16, (kG / 16) / 4), 128, 0, stream>>>(xp, whh1, hc, pre, t);
    k_cell<<<(kB * kH) / 256, 256, 0, stream>>>(pre, cs, hc, h1a, t);
  }

  // FC head (dominant GEMM, bf16 WMMA, L2-resident weights, 2x4 register blocking)
  k_fc<<<dim3((kBT / 16) / 4, (kV / 16) / 16), 256, 0, stream>>>(fcw, h1a, fc_b, out);
}
